// VisionTransformer_10651518894460
// MI455X (gfx1250) — matrix operand order repaired and checked by reading, not compiled, not run on hardware
//
#include <hip/hip_runtime.h>
#include <hip/hip_bf16.h>

// ---------------------------------------------------------------------------
// Problem constants (from the reference)
// ---------------------------------------------------------------------------
#define E_DIM 768
#define HEADS 12
#define DH    64
#define BATCH 64
#define SEQ   197
#define TOK   196
#define TOKP  224                 // TOK padded to a multiple of 32 (K padding)
#define BS    (BATCH * SEQ)       // 12608 rows (all tokens incl. CLS)
#define BH    (BATCH * HEADS)     // 768 attention batches

typedef __attribute__((ext_vector_type(16))) __bf16 v16bf;
typedef __attribute__((ext_vector_type(8)))  __bf16 v8bf;
typedef __attribute__((ext_vector_type(8)))  float  v8f;

// ---------------------------------------------------------------------------
// bf16 <-> f32 helpers (bit-level, RNE)
// ---------------------------------------------------------------------------
__device__ __forceinline__ __bf16 f2bf(float x) {
  union { float f; unsigned u; } v; v.f = x;
  unsigned r = v.u + 0x7fffu + ((v.u >> 16) & 1u);
  unsigned short h = (unsigned short)(r >> 16);
  __bf16 b; __builtin_memcpy(&b, &h, 2); return b;
}
__device__ __forceinline__ float bf2f(__bf16 x) {
  unsigned short h; __builtin_memcpy(&h, &x, 2);
  union { float f; unsigned u; } v; v.u = ((unsigned)h) << 16; return v.f;
}
__device__ __forceinline__ __bf16 bfz() {
  unsigned short h = 0; __bf16 b; __builtin_memcpy(&b, &h, 2); return b;
}

__device__ __forceinline__ float wave_max(float v) {
#pragma unroll
  for (int o = 16; o > 0; o >>= 1) v = fmaxf(v, __shfl_xor(v, o, 32));
  return v;
}
__device__ __forceinline__ float wave_sum(float v) {
#pragma unroll
  for (int o = 16; o > 0; o >>= 1) v += __shfl_xor(v, o, 32);
  return v;
}

#define CAT16 0,1,2,3,4,5,6,7,8,9,10,11,12,13,14,15

// ---------------------------------------------------------------------------
// Generic batched bf16 WMMA GEMM:  C[z] = A[z] (MxK) * Bt[z]^T (KxN) + bias
//   Bt is stored N-major (N rows of K).  REQUIREMENTS:
//     - K is a multiple of 32 (pad operands with zeros in K)
//     - rows of A / Bt are readable at clamped indices (always true here)
//   WMMA fragment layouts per CDNA5 ISA 7.12.2 (wave32):
//     A 16x32: lane l -> row m=l&15, kh=l>>4; halves [k0+kh*8 .. +7] and
//              [k0+16+kh*8 .. +7]
//     B 32x16 from N-major buffer: lane l -> col n=l&15, halves at
//              [k0+kh*8 .. +7] and [k0+16+kh*8 .. +7] of row n
//     C 16x16 f32: elem r -> row r+8*kh, col l&15
//   mode 0: write fp32 (outF, += if accum) and/or bf16 (outH) at z*sC+m*ldC+n
//   mode 1: projection scatter to (bh, t, d) layout (Q/K); CLS row -> outCls
//   mode 2: projection scatter to (bh, d, t) layout, ld TOKP (V/C)
// Block: 128 threads = 4 waves; each wave computes 32 rows x 64 cols.
// ---------------------------------------------------------------------------
__global__ void __launch_bounds__(128) wmma_gemm_bf16(
    const __bf16* __restrict__ A, long ldA, long sA,
    const __bf16* __restrict__ Bt, long ldB, long sB,
    float* __restrict__ outF, __bf16* __restrict__ outH, __bf16* __restrict__ outCls,
    long ldC, long sC, const float* __restrict__ bias,
    int M, int N, int K, int accum, int mode)
{
  int lane = threadIdx.x & 31;
  int wave = threadIdx.x >> 5;
  int nl = lane & 15;
  int kh = lane >> 4;
  long z = blockIdx.z;

  int mBase = blockIdx.y * 128 + wave * 32;  // two 16-row tiles per wave
  int nBase = blockIdx.x * 64;

  const __bf16* Az = A + z * sA;
  const __bf16* Bz = Bt + z * sB;

  // clamped per-lane row pointers (OOB rows read valid memory; their results
  // are discarded by the store guards)
  int rA0 = min(mBase + nl, M - 1);
  int rA1 = min(mBase + 16 + nl, M - 1);
  const __bf16* pA0 = Az + (long)rA0 * ldA + kh * 8;
  const __bf16* pA1 = Az + (long)rA1 * ldA + kh * 8;
  const __bf16* pB[4];
#pragma unroll
  for (int t = 0; t < 4; ++t) {
    int rB = min(nBase + t * 16 + nl, N - 1);
    pB[t] = Bz + (long)rB * ldB + kh * 8;
  }

  v8f zero = {0.f, 0.f, 0.f, 0.f, 0.f, 0.f, 0.f, 0.f};
  v8f acc[2][4];
#pragma unroll
  for (int mt = 0; mt < 2; ++mt)
#pragma unroll
    for (int t = 0; t < 4; ++t) acc[mt][t] = zero;

  for (int k0 = 0; k0 < K; k0 += 32) {
    v8bf a00 = *(const v8bf*)(pA0 + k0);
    v8bf a01 = *(const v8bf*)(pA0 + k0 + 16);
    v8bf a10 = *(const v8bf*)(pA1 + k0);
    v8bf a11 = *(const v8bf*)(pA1 + k0 + 16);
    v16bf a0 = __builtin_shufflevector(a00, a01, CAT16);
    v16bf a1 = __builtin_shufflevector(a10, a11, CAT16);
#pragma unroll
    for (int t = 0; t < 4; ++t) {
      v8bf b0 = *(const v8bf*)(pB[t] + k0);
      v8bf b1 = *(const v8bf*)(pB[t] + k0 + 16);
      v16bf b = __builtin_shufflevector(b0, b1, CAT16);
      acc[0][t] = __builtin_amdgcn_wmma_f32_16x16x32_bf16(
          false, a0, false, b, (short)0, acc[0][t], false, false);
      acc[1][t] = __builtin_amdgcn_wmma_f32_16x16x32_bf16(
          false, a1, false, b, (short)0, acc[1][t], false, false);
    }
  }

  // ---- store epilogue (branches only here) ----
#pragma unroll
  for (int mt = 0; mt < 2; ++mt) {
#pragma unroll
    for (int t = 0; t < 4; ++t) {
      int n = nBase + t * 16 + nl;
      if (n >= N) continue;
      float bb = bias ? bias[n] : 0.f;
#pragma unroll
      for (int r = 0; r < 8; ++r) {
        int m = mBase + mt * 16 + r + 8 * kh;
        if (m >= M) continue;
        float v = acc[mt][t][r] + bb;
        if (mode == 0) {
          long idx = z * sC + (long)m * ldC + n;
          if (outF) outF[idx] = accum ? (outF[idx] + v) : v;
          if (outH) outH[idx] = f2bf(v);
        } else {
          int b = m / SEQ, s = m % SEQ;
          int h = n >> 6, d = n & 63;
          if (s == 0) {
            if (outCls) outCls[(long)b * E_DIM + n] = f2bf(v);
          } else {
            long idx;
            if (mode == 1)  // (bh, t, d) row-major, ld = DH
              idx = (((long)(b * HEADS + h) * TOK) + (s - 1)) * DH + d;
            else            // (bh, d, t) transposed, ld = TOKP
              idx = (((long)(b * HEADS + h) * DH) + d) * TOKP + (s - 1);
            outH[idx] = f2bf(v);
          }
        }
      }
    }
  }
}

// ---------------------------------------------------------------------------
// Elementwise f32 -> bf16 conversion
// ---------------------------------------------------------------------------
__global__ void convert_f32_to_bf16(const float* __restrict__ src,
                                    __bf16* __restrict__ dst, long n) {
  long i = (long)blockIdx.x * blockDim.x + threadIdx.x;
  if (i < n) dst[i] = f2bf(src[i]);
}

// ---------------------------------------------------------------------------
// Zero the K-padding columns [start, ld) of a (rows x ld) bf16 buffer
// ---------------------------------------------------------------------------
__global__ void pad_zero_kernel(__bf16* __restrict__ buf, long rows, int ld, int start) {
  int padw = ld - start;
  long total = rows * padw;
  long i = (long)blockIdx.x * blockDim.x + threadIdx.x;
  if (i >= total) return;
  long r = i / padw;
  int c = start + (int)(i % padw);
  buf[r * ld + c] = bfz();
}

// ---------------------------------------------------------------------------
// LDS-tiled transpose + convert: Wt[c][r] = (bf16)W[r][c]
// ---------------------------------------------------------------------------
__global__ void transpose_to_bf16(const float* __restrict__ W,
                                  __bf16* __restrict__ Wt, int rows, int cols) {
  __shared__ float tile[32][33];
  int c0 = blockIdx.x * 32, r0 = blockIdx.y * 32;
  for (int i = threadIdx.y; i < 32; i += 8) {
    int r = r0 + i, c = c0 + threadIdx.x;
    tile[i][threadIdx.x] = (r < rows && c < cols) ? W[(long)r * cols + c] : 0.f;
  }
  __syncthreads();
  for (int i = threadIdx.y; i < 32; i += 8) {
    int c = c0 + i, r = r0 + threadIdx.x;
    if (c < cols && r < rows) Wt[(long)c * rows + r] = f2bf(tile[threadIdx.x][i]);
  }
}

// ---------------------------------------------------------------------------
// Per-(bh, j): q_norm2 (clipped), cq = c.q_j, ck = c.k_j
// ---------------------------------------------------------------------------
__global__ void norms_kernel(const __bf16* __restrict__ Qh, const __bf16* __restrict__ Kh,
                             const __bf16* __restrict__ clsQ,
                             float* __restrict__ qn2, float* __restrict__ cqv,
                             float* __restrict__ ckv) {
  long t = (long)blockIdx.x * blockDim.x + threadIdx.x;  // bh*196 + j
  if (t >= (long)BH * TOK) return;
  long bh = t / TOK;
  int b = (int)(bh / HEADS), h = (int)(bh % HEADS);
  const __bf16* q = Qh + t * DH;
  const __bf16* k = Kh + t * DH;
  const __bf16* c = clsQ + (long)b * E_DIM + h * DH;
  float qs = 0.f, cq = 0.f, ck = 0.f;
#pragma unroll 8
  for (int d = 0; d < DH; ++d) {
    float qv = bf2f(q[d]);
    float cv = bf2f(c[d]);
    qs += qv * qv;
    cq += cv * qv;
    ck += cv * bf2f(k[d]);
  }
  qn2[t] = fmaxf(qs, 1e-5f);
  cqv[t] = cq;
  ckv[t] = ck;
}

__global__ void cn2_kernel(const __bf16* __restrict__ clsQ, float* __restrict__ cn2) {
  int bh = blockIdx.x * blockDim.x + threadIdx.x;
  if (bh >= BH) return;
  int b = bh / HEADS, h = bh % HEADS;
  const __bf16* c = clsQ + (long)b * E_DIM + h * DH;
  float s = 0.f;
#pragma unroll 8
  for (int d = 0; d < DH; ++d) { float v = bf2f(c[d]); s += v * v; }
  cn2[bh] = fmaxf(s, 1e-5f);
}

// ---------------------------------------------------------------------------
// Wave-per-row dual softmax:
//   l1 = qk / sqrt(Dh*qn2[i])          -> attn_std (fp32 output + bf16 A)
//   l2 = l1 * cq[j] / sqrt(qn2[i]*cn2) -> attn_c2o (bf16 A)
// bf16 A matrices use ld = TOKP; pad columns written as zero (K padding).
// ---------------------------------------------------------------------------
__global__ void __launch_bounds__(256) softmax_rows_kernel(
    const float* __restrict__ QK, const float* __restrict__ qn2,
    const float* __restrict__ cn2, const float* __restrict__ cqv,
    float* __restrict__ attn_out, __bf16* __restrict__ Astd,
    __bf16* __restrict__ Ac2o) {
  int wave = threadIdx.x >> 5, lane = threadIdx.x & 31;
  long row = (long)blockIdx.x * 8 + wave;  // bh*196 + i
  long bh = row / TOK;
  const float* qkrow = QK + row * TOK;
  const float* cqrow = cqv + bh * TOK;
  float qi = qn2[row];
  float s1 = rsqrtf((float)DH * qi);
  float s2 = rsqrtf(qi * cn2[bh]);

  float l1[7], l2[7];
  float m1 = -1e30f, m2 = -1e30f;
#pragma unroll
  for (int it = 0; it < 7; ++it) {
    int j = lane + it * 32;
    if (j < TOK) {
      float l = qkrow[j] * s1;
      float t = l * cqrow[j] * s2;
      l1[it] = l; l2[it] = t;
      m1 = fmaxf(m1, l); m2 = fmaxf(m2, t);
    } else { l1[it] = -1e30f; l2[it] = -1e30f; }
  }
  m1 = wave_max(m1); m2 = wave_max(m2);
  float e1[7], e2[7], s1s = 0.f, s2s = 0.f;
#pragma unroll
  for (int it = 0; it < 7; ++it) {
    int j = lane + it * 32;
    if (j < TOK) {
      e1[it] = __expf(l1[it] - m1); e2[it] = __expf(l2[it] - m2);
      s1s += e1[it]; s2s += e2[it];
    } else { e1[it] = 0.f; e2[it] = 0.f; }
  }
  s1s = wave_sum(s1s); s2s = wave_sum(s2s);
  float r1 = 1.f / s1s, r2 = 1.f / s2s;
#pragma unroll
  for (int it = 0; it < 7; ++it) {
    int j = lane + it * 32;  // j < TOKP always (max 223)
    if (j < TOK) {
      float p1 = e1[it] * r1, p2 = e2[it] * r2;
      attn_out[row * TOK + j] = p1;
      Astd[row * TOKP + j] = f2bf(p1);
      Ac2o[row * TOKP + j] = f2bf(p2);
    } else {
      Astd[row * TOKP + j] = bfz();
      Ac2o[row * TOKP + j] = bfz();
    }
  }
}

// ---------------------------------------------------------------------------
// Wave-per-(bh): attn_o2c = softmax(ck / sqrt(Dh*cn2)); cls_out = attn_o2c @ v
// ---------------------------------------------------------------------------
__global__ void __launch_bounds__(256) o2c_kernel(
    const float* __restrict__ ckv, const float* __restrict__ cn2,
    const __bf16* __restrict__ Vt, float* __restrict__ clsO) {
  int wave = threadIdx.x >> 5, lane = threadIdx.x & 31;
  int bh = blockIdx.x * 8 + wave;
  float s = rsqrtf((float)DH * cn2[bh]);
  const float* ck = ckv + (long)bh * TOK;
  float l[7]; float mx = -1e30f;
#pragma unroll
  for (int it = 0; it < 7; ++it) {
    int j = lane + it * 32;
    l[it] = (j < TOK) ? ck[j] * s : -1e30f;
    mx = fmaxf(mx, l[it]);
  }
  mx = wave_max(mx);
  float p[7]; float sum = 0.f;
#pragma unroll
  for (int it = 0; it < 7; ++it) {
    int j = lane + it * 32;
    p[it] = (j < TOK) ? __expf(l[it] - mx) : 0.f;
    sum += p[it];
  }
  sum = wave_sum(sum);
  float r = 1.f / sum;
  const __bf16* vb = Vt + (long)bh * DH * TOKP;  // (d, j) layout, ld TOKP
  int b = bh / HEADS, h = bh % HEADS;
  for (int d = 0; d < DH; ++d) {
    float acc = 0.f;
#pragma unroll
    for (int it = 0; it < 7; ++it) {
      int j = lane + it * 32;
      if (j < TOK) acc += p[it] * bf2f(vb[(long)d * TOKP + j]);
    }
    acc = wave_sum(acc) * r;
    if (lane == 0) clsO[(long)b * E_DIM + h * DH + d] = acc;
  }
}

// ---------------------------------------------------------------------------
// Assemble (B, 197, 768) activations in bf16: row 0 = cls_out, rows 1.. from
// per-head (bh, t, d) fp32 values.
// ---------------------------------------------------------------------------
__global__ void assemble_kernel(const float* __restrict__ clsO,
                                const float* __restrict__ Valf,
                                __bf16* __restrict__ Ybf) {
  long idx = (long)blockIdx.x * blockDim.x + threadIdx.x;
  if (idx >= (long)BS * E_DIM) return;
  long m = idx / E_DIM;
  int n = (int)(idx % E_DIM);
  int b = (int)(m / SEQ), s = (int)(m % SEQ);
  float v;
  if (s == 0) {
    v = clsO[(long)b * E_DIM + n];
  } else {
    int h = n >> 6, d = n & 63;
    v = Valf[(((long)(b * HEADS + h) * TOK) + (s - 1)) * DH + d];
  }
  Ybf[idx] = f2bf(v);
}

// ---------------------------------------------------------------------------
// Host launcher
// ---------------------------------------------------------------------------
extern "C" void kernel_launch(void* const* d_in, const int* in_sizes, int n_in,
                              void* d_out, int out_size, void* d_ws, size_t ws_size,
                              hipStream_t stream) {
  (void)in_sizes; (void)n_in; (void)out_size; (void)ws_size;
  const float* x  = (const float*)d_in[0];
  const float* Wq = (const float*)d_in[1];
  const float* bq = (const float*)d_in[2];
  const float* Wk = (const float*)d_in[3];
  const float* bk = (const float*)d_in[4];
  const float* Wv = (const float*)d_in[5];
  const float* bv = (const float*)d_in[6];
  const float* Wc = (const float*)d_in[7];
  const float* bc = (const float*)d_in[8];
  const float* Wo = (const float*)d_in[9];
  const float* bo = (const float*)d_in[10];

  float* out      = (float*)d_out;               // (64,197,768)
  float* attn_out = out + (size_t)BS * E_DIM;    // (64,12,196,196)

  char* ws = (char*)d_ws;
  size_t off = 0;
  auto take = [&](size_t bytes) -> void* {
    void* p = ws + off;
    off += (bytes + 255) & ~(size_t)255;
    return p;
  };
  __bf16* Xbf  = (__bf16*)take((size_t)BS * E_DIM * 2);
  __bf16* WqT  = (__bf16*)take((size_t)E_DIM * E_DIM * 2);
  __bf16* WkT  = (__bf16*)take((size_t)E_DIM * E_DIM * 2);
  __bf16* WvT  = (__bf16*)take((size_t)E_DIM * E_DIM * 2);
  __bf16* WcT  = (__bf16*)take((size_t)E_DIM * E_DIM * 2);
  __bf16* WoT  = (__bf16*)take((size_t)E_DIM * E_DIM * 2);
  __bf16* clsQ = (__bf16*)take((size_t)BATCH * E_DIM * 2);
  __bf16* Qh   = (__bf16*)take((size_t)BH * TOK * DH * 2);
  __bf16* Kh   = (__bf16*)take((size_t)BH * TOK * DH * 2);
  __bf16* Vt   = (__bf16*)take((size_t)BH * DH * TOKP * 2);
  __bf16* Ct   = (__bf16*)take((size_t)BH * DH * TOKP * 2);
  float*  QKf  = (float*)take((size_t)BH * TOK * TOK * 4);
  __bf16* Astd = (__bf16*)take((size_t)BH * TOK * TOKP * 2);
  __bf16* Ac2o = (__bf16*)take((size_t)BH * TOK * TOKP * 2);
  float*  qn2  = (float*)take((size_t)BH * TOK * 4);
  float*  cqv  = (float*)take((size_t)BH * TOK * 4);
  float*  ckv  = (float*)take((size_t)BH * TOK * 4);
  float*  cn2  = (float*)take((size_t)BH * 4);
  float*  Valf = (float*)take((size_t)BH * TOK * DH * 4);
  float*  clsO = (float*)take((size_t)BATCH * E_DIM * 4);
  __bf16* Ybf  = (__bf16*)take((size_t)BS * E_DIM * 2);

  // 1) convert x to bf16
  convert_f32_to_bf16<<<dim3((BS * E_DIM) / 256), dim3(256), 0, stream>>>(
      x, Xbf, (long)BS * E_DIM);

  // 2) transpose + convert weights (N-major bf16 for WMMA B fragments)
  dim3 tg(E_DIM / 32, E_DIM / 32), tb(32, 8);
  transpose_to_bf16<<<tg, tb, 0, stream>>>(Wq, WqT, E_DIM, E_DIM);
  transpose_to_bf16<<<tg, tb, 0, stream>>>(Wk, WkT, E_DIM, E_DIM);
  transpose_to_bf16<<<tg, tb, 0, stream>>>(Wv, WvT, E_DIM, E_DIM);
  transpose_to_bf16<<<tg, tb, 0, stream>>>(Wc, WcT, E_DIM, E_DIM);
  transpose_to_bf16<<<tg, tb, 0, stream>>>(Wo, WoT, E_DIM, E_DIM);

  // 2b) zero the K-pad columns of Vt/Ct (j in [196,224))
  {
    long rows = (long)BH * DH;
    long total = rows * (TOKP - TOK);
    int blocks = (int)((total + 255) / 256);
    pad_zero_kernel<<<dim3(blocks), 256, 0, stream>>>(Vt, rows, TOKP, TOK);
    pad_zero_kernel<<<dim3(blocks), 256, 0, stream>>>(Ct, rows, TOKP, TOK);
  }

  // 3) projections with per-head scatter (M=12608, 128 rows per block)
  dim3 pg(E_DIM / 64, (BS + 127) / 128, 1);  // (12, 99)
  wmma_gemm_bf16<<<pg, 128, 0, stream>>>(Xbf, E_DIM, 0, WqT, E_DIM, 0,
      nullptr, Qh, clsQ, 0, 0, bq, BS, E_DIM, E_DIM, 0, 1);
  wmma_gemm_bf16<<<pg, 128, 0, stream>>>(Xbf, E_DIM, 0, WkT, E_DIM, 0,
      nullptr, Kh, nullptr, 0, 0, bk, BS, E_DIM, E_DIM, 0, 1);
  wmma_gemm_bf16<<<pg, 128, 0, stream>>>(Xbf, E_DIM, 0, WvT, E_DIM, 0,
      nullptr, Vt, nullptr, 0, 0, bv, BS, E_DIM, E_DIM, 0, 2);
  wmma_gemm_bf16<<<pg, 128, 0, stream>>>(Xbf, E_DIM, 0, WcT, E_DIM, 0,
      nullptr, Ct, nullptr, 0, 0, bc, BS, E_DIM, E_DIM, 0, 2);

  // 4) norms, cq, ck, c_norm2
  norms_kernel<<<dim3((BH * TOK) / 256), 256, 0, stream>>>(Qh, Kh, clsQ, qn2, cqv, ckv);
  cn2_kernel<<<dim3(3), 256, 0, stream>>>(clsQ, cn2);

  // 5) qk = q @ k^T  (batched over bh; M=N=196, K=64)
  dim3 qkg(4, 2, BH);
  wmma_gemm_bf16<<<qkg, 128, 0, stream>>>(Qh, DH, (long)TOK * DH, Kh, DH, (long)TOK * DH,
      QKf, nullptr, nullptr, TOK, (long)TOK * TOK, nullptr, TOK, TOK, DH, 0, 0);

  // 6) dual softmax -> attn_std (fp32 out + bf16 A), attn_c2o (bf16 A)
  softmax_rows_kernel<<<dim3((BH * TOK) / 8), 256, 0, stream>>>(
      QKf, qn2, cn2, cqv, attn_out, Astd, Ac2o);

  // 7) values = attn_std @ v + attn_c2o @ cls_proj  (K padded to 224)
  dim3 vg(1, 2, BH);
  wmma_gemm_bf16<<<vg, 128, 0, stream>>>(Astd, TOKP, (long)TOK * TOKP, Vt, TOKP, (long)DH * TOKP,
      Valf, nullptr, nullptr, DH, (long)TOK * DH, nullptr, TOK, DH, TOKP, 0, 0);
  wmma_gemm_bf16<<<vg, 128, 0, stream>>>(Ac2o, TOKP, (long)TOK * TOKP, Ct, TOKP, (long)DH * TOKP,
      Valf, nullptr, nullptr, DH, (long)TOK * DH, nullptr, TOK, DH, TOKP, 1, 0);

  // 8) CLS attention row
  o2c_kernel<<<dim3(BH / 8), 256, 0, stream>>>(ckv, cn2, Vt, clsO);

  // 9) assemble (B,197,768) bf16
  assemble_kernel<<<dim3((BS * E_DIM) / 256), 256, 0, stream>>>(clsO, Valf, Ybf);

  // 10) out = Y @ Wo + bo  (fp32 straight into d_out)
  wmma_gemm_bf16<<<pg, 128, 0, stream>>>(Ybf, E_DIM, 0, WoT, E_DIM, 0,
      out, nullptr, nullptr, E_DIM, 0, bo, BS, E_DIM, E_DIM, 0, 0);
}
